// TFreeEncoder_78451872629346
// MI455X (gfx1250) — compile-verified, matrix operand order checked by reading, not run on hardware
//
#include <hip/hip_runtime.h>
#include <hip/hip_bf16.h>

// ---------------------------------------------------------------------------
// TFree encoder on MI455X (gfx1250, wave32, WMMA).
// All matmuls via v_wmma_f32_16x16x32_bf16 (bf16 operands, f32 accum).
// ---------------------------------------------------------------------------

typedef __bf16 bf16_t;
typedef __bf16 v16bf __attribute__((ext_vector_type(16)));
typedef __bf16 v8bf  __attribute__((ext_vector_type(8)));
typedef float  v8f   __attribute__((ext_vector_type(8)));

// Global-address-space fragment pointer: keeps panel streams on
// global_load_b128 (LOADcnt only) instead of flat (LOADcnt+DScnt).
typedef const v16bf __attribute__((address_space(1)))* gfragp;

#define N_    8192
#define M_    16
#define H_    512
#define IH_   128
#define HH_   256
#define G4_   1024          // 4*HH (gates)
#define KC_   384           // IH + HH (concat K of fused input+recurrent GEMM)
#define KT_R  12            // KC_/32 k-tiles (recurrence)
#define NT_R  64            // G4_/16 n-tiles (recurrence)
#define KT_P  16            // H_/32  k-tiles (projection)
#define NT_P  32            // H_/16  n-tiles (projection)
#define MB_   32            // words per LSTM block (2 WMMA M-tiles share each B)

__device__ __forceinline__ float sigmoidf_(float x) {
    return 1.0f / (1.0f + __expf(-x));
}

__device__ __forceinline__ v8f wmma_bf16(v16bf a, v16bf b, v8f c) {
    // D = A(16x32 bf16) * B(32x16 bf16) + C(16x16 f32)
    return __builtin_amdgcn_wmma_f32_16x16x32_bf16(
        /*neg_a=*/false, a, /*neg_b=*/false, b,
        /*c_mod=*/(short)0, c, /*reuse_a=*/false, /*reuse_b=*/false);
}

// Load one A-operand fragment from a bf16 LDS row.
// ISA layout: lanes 0-15 hold K = {kt*32 + 0..7, kt*32 + 16..23},
//             lanes 16-31 add +8.  base16B must point at (row, kt*32 + kofs).
__device__ __forceinline__ v16bf lds_frag(const bf16_t* base16B) {
    union { v16bf v; v8bf h[2]; } u;
    u.h[0] = *reinterpret_cast<const v8bf*>(base16B);       // k .. k+7
    u.h[1] = *reinterpret_cast<const v8bf*>(base16B + 16);  // k+16 .. k+23
    return u.v;
}

// ---------------------------------------------------------------------------
// Kernel 0a: pack W_cat = [w_ih^T ; w_hh^T] (384 x 1024) into B-fragment order:
//   dst[((kt*NT_R + nt)*32 + lane)*16 + j] ; B element (k, n) = W[n][k]
// ---------------------------------------------------------------------------
__global__ void tfree_pack_wcat(const float* __restrict__ w_ih,
                                const float* __restrict__ w_hh,
                                bf16_t* __restrict__ dst) {
    int idx = blockIdx.x * blockDim.x + threadIdx.x;
    if (idx >= KT_R * NT_R * 32 * 16) return;
    int j    = idx & 15;
    int lane = (idx >> 4) & 31;
    int nt   = (idx >> 9) & 63;
    int kt   = idx >> 15;
    int n = nt * 16 + (lane & 15);
    int k = kt * 32 + (j < 8 ? j : j + 8) + (lane >= 16 ? 8 : 0);
    float v = (k < IH_) ? w_ih[n * IH_ + k] : w_hh[n * HH_ + (k - IH_)];
    dst[idx] = (bf16_t)v;
}

// Kernel 0b: pack proj_w^T (512 x 512, B element (k,o) = proj_w[o][k]).
__global__ void tfree_pack_proj(const float* __restrict__ proj_w,
                                bf16_t* __restrict__ dst) {
    int idx = blockIdx.x * blockDim.x + threadIdx.x;
    if (idx >= KT_P * NT_P * 32 * 16) return;
    int j    = idx & 15;
    int lane = (idx >> 4) & 31;
    int nt   = (idx >> 9) & 31;
    int kt   = idx >> 14;
    int n = nt * 16 + (lane & 15);
    int k = kt * 32 + (j < 8 ? j : j + 8) + (lane >= 16 ? 8 : 0);
    dst[idx] = (bf16_t)proj_w[n * H_ + k];
}

// Kernel 0c: combined LSTM biases b_ih + b_hh, both directions.
__global__ void tfree_pack_bias(const float* bi_f, const float* bh_f,
                                const float* bi_b, const float* bh_b,
                                float* bias_f, float* bias_b) {
    int i = blockIdx.x * blockDim.x + threadIdx.x;
    if (i < G4_) {
        bias_f[i] = bi_f[i] + bh_f[i];
        bias_b[i] = bi_b[i] + bh_b[i];
    }
}

// ---------------------------------------------------------------------------
// Kernel 1: fused BiLSTM recurrence.  grid = (N/32, 2 dirs), block = 256
// (8 waves).  Per block: 32 words as two M=16 tiles, so each L2-resident
// B-fragment of the 768 KB W_cat panel feeds two WMMAs.  Per step:
//   g(32x1024) = [x_t | h] (32x384) @ W_cat + bias
// Wave wv owns hidden units u in [wv*32, wv*32+32) across ALL four gates,
// so gate math + c state stay in registers; h round-trips LDS as bf16
// (it is exactly next step's A operand).  The weight-panel pointer is
// laundered through an empty asm each step (blocks LICM from hoisting the
// 96 fragment loads -> scratch spills) and rebuilt as an address_space(1)
// pointer (keeps the loads on global_load_b128, not flat).
// xall is staged [t][w][f] so both M-tiles' A-fragments sit within the
// 16-bit DS immediate-offset range of one base address.
// ---------------------------------------------------------------------------
__global__ __launch_bounds__(256)
void tfree_lstm(const int* __restrict__ char_ids,
                const int* __restrict__ char_lens,
                const float* __restrict__ char_emb,
                const bf16_t* __restrict__ wcat_f,
                const bf16_t* __restrict__ wcat_b,
                const float* __restrict__ bias_f,
                const float* __restrict__ bias_b,
                bf16_t* __restrict__ char_hidden /* [N][512] bf16 */) {
    extern __shared__ char smem[];
    bf16_t* xall = (bf16_t*)smem;                              // [16t][32w][128] = 128 KB
    bf16_t* hbf  = (bf16_t*)(smem + MB_ * 16 * IH_ * 2);       // [32w][256]      =  16 KB
    int*    lens_s = (int*)(smem + MB_ * 16 * IH_ * 2 + MB_ * HH_ * 2); // 32 ints

    const int blk  = blockIdx.x;
    const int dir  = blockIdx.y;
    const bf16_t* wcat = dir ? wcat_b : wcat_f;
    const float*  bias = dir ? bias_b : bias_f;
    const int tid  = threadIdx.x;
    const int lane = tid & 31;
    const int wv   = tid >> 5;

    if (tid < MB_) lens_s[tid] = char_lens[blk * MB_ + tid];
    __syncthreads();  // lens needed for backward-direction gather below

    // Stage char embeddings for all 16 steps, layout [t][w][f]
    // (bwd: length-reversed gather).
    for (int i = tid; i < MB_ * 16 * IH_; i += 256) {
        int f = i & 127;
        int w = (i >> 7) & 31;
        int t = i >> 12;
        int len = lens_s[w];
        int tt = t;
        if (dir) { tt = len - 1 - t; tt = tt < 0 ? 0 : (tt > 15 ? 15 : tt); }
        int cid = char_ids[(blk * MB_ + w) * M_ + tt];
        xall[i] = (bf16_t)char_emb[cid * IH_ + f];
    }
    for (int i = tid; i < MB_ * HH_; i += 256) hbf[i] = (bf16_t)0.0f;
    __syncthreads();

    const int mrow  = lane & 15;            // word row within an M-tile
    const int kofs  = (lane >= 16) ? 8 : 0;
    const int rbase = (lane >= 16) ? 8 : 0; // C/D row offset for this lane half

    // Per-lane bias (same value for all 8 accumulator rows), held in regs.
    float breg[4][2];
#pragma unroll
    for (int G = 0; G < 4; ++G)
#pragma unroll
        for (int s = 0; s < 2; ++s)
            breg[G][s] = bias[(G * 16 + wv * 2 + s) * 16 + (lane & 15)];

    float hreg[2][2][8], creg[2][2][8];     // [mtile][sub][row]
#pragma unroll
    for (int mt = 0; mt < 2; ++mt)
#pragma unroll
        for (int s = 0; s < 2; ++s)
#pragma unroll
            for (int r = 0; r < 8; ++r) {
                hreg[mt][s][r] = 0.0f;
                creg[mt][s][r] = 0.0f;
            }

    unsigned long long wbits = (unsigned long long)wcat;

#pragma unroll 1
    for (int t = 0; t < M_; ++t) {
        // Launder the panel pointer: blocks LICM from hoisting B loads.
        asm volatile("" : "+s"(wbits));
        gfragp wfrag = (gfragp)wbits;   // global address space -> global_load

        // ---- accumulators = bias ----
        v8f acc[4][2][2];
#pragma unroll
        for (int G = 0; G < 4; ++G)
#pragma unroll
            for (int s = 0; s < 2; ++s) {
                float b = breg[G][s];
                v8f a = {b, b, b, b, b, b, b, b};
                acc[G][s][0] = a;
                acc[G][s][1] = a;
            }

        // ---- GEMM: [x_t | h] @ W_cat, 12 k-tiles, 2 M-tiles per B ----
#pragma unroll
        for (int kt = 0; kt < KT_R; ++kt) {
            const bf16_t* a0;
            int mstride;                 // elements between the two M-tiles
            if (kt < 4) {
                a0 = xall + (t * MB_ + mrow) * IH_ + kt * 32 + kofs;
                mstride = 16 * IH_;      // +4096 B: fits DS offset
            } else {
                a0 = hbf + mrow * HH_ + (kt - 4) * 32 + kofs;
                mstride = 16 * HH_;      // +8192 B: fits DS offset
            }
            v16bf af0 = lds_frag(a0);
            v16bf af1 = lds_frag(a0 + mstride);
#pragma unroll
            for (int G = 0; G < 4; ++G)
#pragma unroll
                for (int s = 0; s < 2; ++s) {
                    int nt = G * 16 + wv * 2 + s;
                    v16bf bfr = wfrag[(kt * NT_R + nt) * 32 + lane];
                    acc[G][s][0] = wmma_bf16(af0, bfr, acc[G][s][0]);
                    acc[G][s][1] = wmma_bf16(af1, bfr, acc[G][s][1]);
                }
        }
        __syncthreads();  // all waves done reading hbf for step t

        // ---- gates + state update (registers only; write h as bf16) ----
#pragma unroll
        for (int mt = 0; mt < 2; ++mt)
#pragma unroll
            for (int s = 0; s < 2; ++s) {
                int u = wv * 32 + s * 16 + (lane & 15);
#pragma unroll
                for (int r = 0; r < 8; ++r) {
                    int row = mt * 16 + r + rbase;
                    float iv = sigmoidf_(acc[0][s][mt][r]);
                    float fv = sigmoidf_(acc[1][s][mt][r]);
                    float gv = tanhf(acc[2][s][mt][r]);
                    float ov = sigmoidf_(acc[3][s][mt][r]);
                    float cn = fv * creg[mt][s][r] + iv * gv;
                    float hn = ov * tanhf(cn);
                    bool mk = t < lens_s[row];
                    creg[mt][s][r] = mk ? cn : creg[mt][s][r];
                    hreg[mt][s][r] = mk ? hn : hreg[mt][s][r];
                    hbf[row * HH_ + u] = (bf16_t)hreg[mt][s][r];
                }
            }
        __syncthreads();  // hbf ready for step t+1
    }

    // ---- final hidden -> char_hidden[:, dir*256 : dir*256+256] (bf16) ----
#pragma unroll
    for (int mt = 0; mt < 2; ++mt)
#pragma unroll
        for (int s = 0; s < 2; ++s) {
            int u = wv * 32 + s * 16 + (lane & 15);
#pragma unroll
            for (int r = 0; r < 8; ++r) {
                int row = mt * 16 + r + rbase;
                char_hidden[(blk * MB_ + row) * H_ + dir * HH_ + u] =
                    (bf16_t)hreg[mt][s][r];
            }
        }
}

// ---------------------------------------------------------------------------
// Kernel 2: OOV select + projection.  out = sel @ proj_w^T + proj_b.
// grid = N/16 blocks; block = 256 (8 waves, 4 n-tiles each); K = 512.
// ---------------------------------------------------------------------------
__global__ __launch_bounds__(256)
void tfree_project(const int* __restrict__ word_ids,
                   const int* __restrict__ is_oov,
                   const float* __restrict__ word_emb,
                   const bf16_t* __restrict__ char_hidden,
                   const bf16_t* __restrict__ projpk,
                   const float* __restrict__ proj_b,
                   float* __restrict__ out) {
    __shared__ bf16_t atile[16 * H_];  // 16 KB

    const int blk  = blockIdx.x;
    const int tid  = threadIdx.x;
    const int lane = tid & 31;
    const int wv   = tid >> 5;

    // Stage A tile with the OOV select fused in.
    for (int i = tid; i < 16 * H_; i += 256) {
        int k = i & (H_ - 1);
        int m = i >> 9;
        int n = blk * 16 + m;
        bf16_t v;
        if (is_oov[n] > 0) v = char_hidden[n * H_ + k];
        else               v = (bf16_t)word_emb[(long)word_ids[n] * H_ + k];
        atile[i] = v;
    }
    __syncthreads();

    const int mrow = lane & 15;
    const int kofs = (lane >= 16) ? 8 : 0;
    const int rbase = (lane >= 16) ? 8 : 0;
    gfragp pfrag = (gfragp)(unsigned long long)projpk;

    v8f acc[4];
#pragma unroll
    for (int s = 0; s < 4; ++s) {
        v8f z = {0.f, 0.f, 0.f, 0.f, 0.f, 0.f, 0.f, 0.f};
        acc[s] = z;
    }

#pragma unroll 2
    for (int kt = 0; kt < KT_P; ++kt) {
        v16bf afr = lds_frag(atile + mrow * H_ + kt * 32 + kofs);
#pragma unroll
        for (int s = 0; s < 4; ++s) {
            int nt = wv * 4 + s;
            v16bf bfr = pfrag[(kt * NT_P + nt) * 32 + lane];
            acc[s] = wmma_bf16(afr, bfr, acc[s]);
        }
    }

#pragma unroll
    for (int s = 0; s < 4; ++s) {
        int nt  = wv * 4 + s;
        int col = nt * 16 + (lane & 15);
        float pb = proj_b[col];
#pragma unroll
        for (int r = 0; r < 8; ++r) {
            out[(blk * 16 + r + rbase) * H_ + col] = acc[s][r] + pb;
        }
    }
}

// ---------------------------------------------------------------------------
extern "C" void kernel_launch(void* const* d_in, const int* in_sizes, int n_in,
                              void* d_out, int out_size, void* d_ws, size_t ws_size,
                              hipStream_t stream) {
    const int*   word_ids  = (const int*)d_in[0];
    const int*   is_oov    = (const int*)d_in[1];
    const int*   char_ids  = (const int*)d_in[2];
    const int*   char_lens = (const int*)d_in[3];
    const float* word_emb  = (const float*)d_in[4];
    const float* char_emb  = (const float*)d_in[5];
    const float* w_ih_f    = (const float*)d_in[6];
    const float* w_hh_f    = (const float*)d_in[7];
    const float* b_ih_f    = (const float*)d_in[8];
    const float* b_hh_f    = (const float*)d_in[9];
    const float* w_ih_b    = (const float*)d_in[10];
    const float* w_hh_b    = (const float*)d_in[11];
    const float* b_ih_b    = (const float*)d_in[12];
    const float* b_hh_b    = (const float*)d_in[13];
    const float* proj_w    = (const float*)d_in[14];
    const float* proj_b    = (const float*)d_in[15];
    float* out = (float*)d_out;

    // Workspace layout (all offsets 256B-aligned); total ~10.5 MB.
    char* ws = (char*)d_ws;
    const size_t SZ_WCAT = (size_t)KT_R * NT_R * 32 * 16 * sizeof(bf16_t); // 786432
    const size_t SZ_PROJ = (size_t)KT_P * NT_P * 32 * 16 * sizeof(bf16_t); // 524288
    bf16_t* wcat_f  = (bf16_t*)(ws);
    bf16_t* wcat_b  = (bf16_t*)(ws + SZ_WCAT);
    bf16_t* projpk  = (bf16_t*)(ws + 2 * SZ_WCAT);
    float*  bias_f  = (float*)(ws + 2 * SZ_WCAT + SZ_PROJ);
    float*  bias_b  = (float*)(ws + 2 * SZ_WCAT + SZ_PROJ + 4096);
    bf16_t* char_hd = (bf16_t*)(ws + 2 * SZ_WCAT + SZ_PROJ + 8192);
    // char_hd: N_*H_*2 = 8 MB

    // 0) pack weights into WMMA fragment order (L2-resident panels)
    tfree_pack_wcat<<<(KT_R * NT_R * 32 * 16 + 255) / 256, 256, 0, stream>>>(
        w_ih_f, w_hh_f, wcat_f);
    tfree_pack_wcat<<<(KT_R * NT_R * 32 * 16 + 255) / 256, 256, 0, stream>>>(
        w_ih_b, w_hh_b, wcat_b);
    tfree_pack_proj<<<(KT_P * NT_P * 32 * 16 + 255) / 256, 256, 0, stream>>>(
        proj_w, projpk);
    tfree_pack_bias<<<(G4_ + 255) / 256, 256, 0, stream>>>(
        b_ih_f, b_hh_f, b_ih_b, b_hh_b, bias_f, bias_b);

    // 1) fused BiLSTM recurrence (32 words/block, dir in grid.y)
    const size_t lstm_smem = (size_t)MB_ * 16 * IH_ * 2   // xall
                           + (size_t)MB_ * HH_ * 2        // hbf
                           + (size_t)MB_ * 4;             // lens
    tfree_lstm<<<dim3(N_ / MB_, 2), 256, lstm_smem, stream>>>(
        char_ids, char_lens, char_emb, wcat_f, wcat_b, bias_f, bias_b, char_hd);

    // 2) OOV select + projection
    tfree_project<<<N_ / 16, 256, 0, stream>>>(
        word_ids, is_oov, word_emb, char_hd, projpk, proj_b, out);
}
